// logsparseAttention_19877108646133
// MI455X (gfx1250) — matrix operand order repaired and checked by reading, not compiled, not run on hardware
//
#include <hip/hip_runtime.h>

// logsparse attention, L=2048, sub_len=5: the log-sparse branch condition
// (L//5)*2*ceil(log2 5) = 2454 > index holds for every row, so the mask
// degenerates to pure causal. Fused flash-attention-2, bf16 WMMA, f32 acc.
//
//   pass 1: K fp32 -> bf16            (d_ws, [B,L,H,D])
//   pass 2: V fp32 -> bf16 TRANSPOSED (d_ws, [B,H,D,L]) via LDS tiles
//   pass 3: flash kernel; 64-key blocks staged to LDS by the Tensor Data
//           Mover (double buffered, s_wait_tensorcnt). Steady-state loop has
//           no masking; only the diagonal block instantiates masked code.

#define Bb   4
#define Lh   2048
#define Hh   16
#define Dh   64
#define BQ   64     // query rows per workgroup
#define NW   4      // waves per workgroup
#define KB   64     // key block size

typedef __attribute__((ext_vector_type(16))) __bf16 v16bf;
typedef __attribute__((ext_vector_type(8)))  float  v8f;
typedef __attribute__((ext_vector_type(4)))  unsigned v4u;
typedef __attribute__((ext_vector_type(8)))  int      v8i;
typedef __attribute__((ext_vector_type(4)))  int      v4i;

union AF { v16bf v; __bf16 e[16]; unsigned u[8]; uint4 q[2]; };
union PK { unsigned u; __bf16 e[2]; };

__device__ __forceinline__ v8f wmma_bf16(const AF& a, const AF& b, v8f c) {
    return __builtin_amdgcn_wmma_f32_16x16x32_bf16(false, a.v, false, b.v,
                                                   (short)0, c, false, false);
}

// ---------------- pass 1: K fp32 -> bf16 (same layout) ----------------
__global__ __launch_bounds__(256) void cvt_k_bf16(const float* __restrict__ K,
                                                  unsigned* __restrict__ Kb) {
    const int i = blockIdx.x * 256 + threadIdx.x;     // pair index
    PK pk;
    pk.e[0] = (__bf16)K[2 * i];
    pk.e[1] = (__bf16)K[2 * i + 1];
    Kb[i] = pk.u;
}

// ------- pass 2: V fp32 [B,L,H,D] -> bf16 transposed [B,H,D,L] --------
__global__ __launch_bounds__(256) void cvt_v_bf16_t(const float* __restrict__ V,
                                                    unsigned* __restrict__ Vt) {
    __shared__ float tile[64][65];
    const int nl = Lh / 64;
    const int l0 = (blockIdx.x % nl) * 64;
    const int bh = blockIdx.x / nl;
    const int h = bh % Hh, b = bh / Hh;
    const int tid = threadIdx.x;
    #pragma unroll
    for (int i = 0; i < 16; ++i) {
        const int idx = tid + 256 * i;
        const int row = idx >> 6, col = idx & 63;
        tile[row][col] = V[(((size_t)b * Lh + l0 + row) * Hh + h) * Dh + col];
    }
    __syncthreads();
    #pragma unroll
    for (int i = 0; i < 8; ++i) {
        const int idx = tid + 256 * i;
        const int d = idx >> 5, l2 = idx & 31;
        PK pk;
        pk.e[0] = (__bf16)tile[2 * l2][d];
        pk.e[1] = (__bf16)tile[2 * l2 + 1][d];
        Vt[(((size_t)b * Hh + h) * Dh + d) * (Lh / 2) + (l0 >> 1) + l2] = pk.u;
    }
}

// ------------------------- TDM descriptor helpers ----------------------
__device__ __forceinline__ v4u tdm_g0(unsigned lds_off, unsigned long long ga) {
    v4u g0;
    g0[0] = 1u;                                  // count=1, user descriptor
    g0[1] = lds_off;                             // lds_addr
    g0[2] = (unsigned)ga;                        // global_addr[31:0]
    g0[3] = (unsigned)(ga >> 32) | 0x80000000u;  // addr[56:32] | type=2
    return g0;
}

__device__ __forceinline__ v8i tdm_g1(unsigned tensor_d0, unsigned tile_d0,
                                      unsigned tile_d1, unsigned stride_d0) {
    v8i g1;
    g1[0] = (int)(1u << 16);                           // data_size=1 (2B), mask=0
    g1[1] = (int)((tensor_d0 & 0xFFFFu) << 16);        // tensor_dim0 lo16
    g1[2] = (int)(tensor_d0 >> 16);                    // dim0 hi, dim1 lo16 = 0
    g1[3] = (int)((1u << 4) | (tile_d0 << 16));        // tensor_dim1 = 1<<20 | tile_dim0
    g1[4] = (int)(tile_d1 & 0xFFFFu);                  // tile_dim1; tile_dim2=0
    g1[5] = (int)stride_d0;                            // tensor_dim0_stride lo32
    g1[6] = 0;
    g1[7] = 0;
    return g1;
}

__device__ __forceinline__ void tdm_issue(v4u g0, v8i g1) {
    const v4i z = {0, 0, 0, 0};
#if defined(__clang_major__) && __clang_major__ >= 23
    const v8i z8 = {0, 0, 0, 0, 0, 0, 0, 0};
    __builtin_amdgcn_tensor_load_to_lds(g0, g1, z, z, z8, 0);
#else
    __builtin_amdgcn_tensor_load_to_lds(g0, g1, z, z, 0);
#endif
}

// ------------------ one 64-key block of flash attention ----------------
template <bool MASKED>
__device__ __forceinline__ void process_block(
    const __bf16 (&Kc)[KB][Dh], const __bf16 (&Vc)[Dh][KB], __bf16 (&Pw)[16][KB],
    const AF (&aq)[2], v8f (&acc)[4], float (&m)[8], float (&l)[8],
    int kv0, int qrow0, int qrow0_s, int ln16, int hi)
{
    float sc[4][8], mx[8];
    #pragma unroll
    for (int r = 0; r < 8; ++r) mx[r] = -3.0e38f;

    // ---- S = Q K^T : 16 rows x 64 keys, four 16x16 subtiles ----
    #pragma unroll
    for (int t = 0; t < 4; ++t) {
        bool act = true;
        if (MASKED) act = (kv0 + 16 * t) <= (qrow0_s + 15);   // scalar branch
        if (act) {
            AF bk0, bk1;
            const uint4* kp0 = (const uint4*)&Kc[16 * t + ln16][8 * hi];
            const uint4* kp1 = (const uint4*)&Kc[16 * t + ln16][32 + 8 * hi];
            bk0.q[0] = kp0[0]; bk0.q[1] = kp0[2];
            bk1.q[0] = kp1[0]; bk1.q[1] = kp1[2];
            v8f c = (v8f){0.f,0.f,0.f,0.f,0.f,0.f,0.f,0.f};
            c = wmma_bf16(aq[0], bk0, c);
            c = wmma_bf16(aq[1], bk1, c);
            if (MASKED) {
                const int col = kv0 + 16 * t + ln16;
                #pragma unroll
                for (int r = 0; r < 8; ++r) {
                    const int rowg = qrow0 + r + 8 * hi;
                    sc[t][r] = (col > rowg) ? -1.0e9f : c[r];
                    mx[r] = fmaxf(mx[r], sc[t][r]);
                }
            } else {
                #pragma unroll
                for (int r = 0; r < 8; ++r) {
                    sc[t][r] = c[r];
                    mx[r] = fmaxf(mx[r], sc[t][r]);
                }
            }
        } else {
            #pragma unroll
            for (int r = 0; r < 8; ++r) sc[t][r] = -1.0e9f;   // -> weight 0
        }
    }

    // ---- online softmax (exp2 domain; log2e folded into Q) ----
    #pragma unroll
    for (int off = 1; off < 16; off <<= 1)
        #pragma unroll
        for (int r = 0; r < 8; ++r)
            mx[r] = fmaxf(mx[r], __shfl_xor(mx[r], off, 16));

    float scl[8], rs[8];
    #pragma unroll
    for (int r = 0; r < 8; ++r) {
        const float mn = fmaxf(m[r], mx[r]);
        scl[r] = exp2f(m[r] - mn);
        m[r] = mn;
        rs[r] = 0.f;
    }
    #pragma unroll
    for (int t = 0; t < 4; ++t)
        #pragma unroll
        for (int r = 0; r < 8; ++r) {
            const float pe = exp2f(sc[t][r] - m[r]);   // p overwrites sc
            sc[t][r] = pe;
            rs[r] += pe;
        }
    #pragma unroll
    for (int off = 1; off < 16; off <<= 1)
        #pragma unroll
        for (int r = 0; r < 8; ++r)
            rs[r] += __shfl_xor(rs[r], off, 16);
    #pragma unroll
    for (int r = 0; r < 8; ++r) l[r] = l[r] * scl[r] + rs[r];
    #pragma unroll
    for (int j = 0; j < 4; ++j)
        #pragma unroll
        for (int r = 0; r < 8; ++r) acc[j][r] *= scl[r];

    // ---- C-layout P -> LDS -> A-layout (intra-wave transpose) ----
    #pragma unroll
    for (int t = 0; t < 4; ++t)
        #pragma unroll
        for (int r = 0; r < 8; ++r)
            Pw[r + 8 * hi][16 * t + ln16] = (__bf16)sc[t][r];
    asm volatile("s_wait_dscnt 0" ::: "memory");

    AF ap0;
    ap0.q[0] = *(const uint4*)&Pw[ln16][8 * hi];
    ap0.q[1] = *(const uint4*)&Pw[ln16][16 + 8 * hi];

    bool act32 = true;
    if (MASKED) act32 = (kv0 + 32) <= (qrow0_s + 15);

    // ---- O += P V ----
    #pragma unroll
    for (int j = 0; j < 4; ++j) {
        AF bv0;
        const uint4* vp0 = (const uint4*)&Vc[16 * j + ln16][8 * hi];
        bv0.q[0] = vp0[0]; bv0.q[1] = vp0[2];
        acc[j] = wmma_bf16(ap0, bv0, acc[j]);
    }
    if (act32) {
        AF ap1;
        ap1.q[0] = *(const uint4*)&Pw[ln16][32 + 8 * hi];
        ap1.q[1] = *(const uint4*)&Pw[ln16][48 + 8 * hi];
        #pragma unroll
        for (int j = 0; j < 4; ++j) {
            AF bv1;
            const uint4* vp1 = (const uint4*)&Vc[16 * j + ln16][32 + 8 * hi];
            bv1.q[0] = vp1[0]; bv1.q[1] = vp1[2];
            acc[j] = wmma_bf16(ap1, bv1, acc[j]);
        }
    }
}

// --------------------------- pass 3: flash ----------------------------
__global__ __launch_bounds__(128) void logsparse_flash_causal(
    const float* __restrict__ Q, const __bf16* __restrict__ Kbf,
    const __bf16* __restrict__ Vtb, float* __restrict__ Out)
{
    __shared__ __bf16 Kt[2][KB][Dh];     // 2 x 8 KB, key-major
    __shared__ __bf16 Vt[2][Dh][KB];     // 2 x 8 KB, d-major (transposed)
    __shared__ __bf16 Pb[NW][16][KB];    // 8 KB P bounce

    const int tid  = threadIdx.x;
    const int wave = tid >> 5;
    const int lane = tid & 31;
    const int ln16 = lane & 15;
    const int hi   = lane >> 4;

    const int bh      = blockIdx.y;
    const int b       = bh / Hh;
    const int h       = bh % Hh;
    const int qrow0   = blockIdx.x * BQ + wave * 16;
    const int qrow0_s = __builtin_amdgcn_readfirstlane(qrow0);
    const bool issuer = (__builtin_amdgcn_readfirstlane(wave) == 0);

    // Q -> A frags; fold log2(e) so softmax runs in native exp2 domain
    AF aq[2];
    {
        const float* qb = Q + (((size_t)b * Lh + (qrow0 + ln16)) * Hh + h) * Dh;
        const float L2E = 1.44269504088896340736f;
        #pragma unroll
        for (int f = 0; f < 2; ++f) {
            const float4 a0 = *(const float4*)&qb[32 * f + 8 * hi];
            const float4 a1 = *(const float4*)&qb[32 * f + 8 * hi + 4];
            const float4 b0 = *(const float4*)&qb[32 * f + 16 + 8 * hi];
            const float4 b1 = *(const float4*)&qb[32 * f + 16 + 8 * hi + 4];
            aq[f].e[0]  = (__bf16)(a0.x * L2E); aq[f].e[1]  = (__bf16)(a0.y * L2E);
            aq[f].e[2]  = (__bf16)(a0.z * L2E); aq[f].e[3]  = (__bf16)(a0.w * L2E);
            aq[f].e[4]  = (__bf16)(a1.x * L2E); aq[f].e[5]  = (__bf16)(a1.y * L2E);
            aq[f].e[6]  = (__bf16)(a1.z * L2E); aq[f].e[7]  = (__bf16)(a1.w * L2E);
            aq[f].e[8]  = (__bf16)(b0.x * L2E); aq[f].e[9]  = (__bf16)(b0.y * L2E);
            aq[f].e[10] = (__bf16)(b0.z * L2E); aq[f].e[11] = (__bf16)(b0.w * L2E);
            aq[f].e[12] = (__bf16)(b1.x * L2E); aq[f].e[13] = (__bf16)(b1.y * L2E);
            aq[f].e[14] = (__bf16)(b1.z * L2E); aq[f].e[15] = (__bf16)(b1.w * L2E);
        }
    }

    v8f acc[4];
    float m[8], l[8];
    #pragma unroll
    for (int j = 0; j < 4; ++j) acc[j] = (v8f){0.f,0.f,0.f,0.f,0.f,0.f,0.f,0.f};
    #pragma unroll
    for (int r = 0; r < 8; ++r) { m[r] = -3.0e38f; l[r] = 0.f; }

    const int nkb = blockIdx.x + 1;          // causal: keys < (bx+1)*64
    unsigned long long gaK =
        (unsigned long long)(uintptr_t)(Kbf + ((size_t)b * Lh) * (Hh * Dh) + (size_t)h * Dh);
    unsigned long long gaV =
        (unsigned long long)(uintptr_t)(Vtb + (((size_t)b * Hh + h) * Dh) * Lh);
    const v8i g1k = tdm_g1(Dh, Dh, KB, Hh * Dh);   // K: 64x64 tile, row stride 1024
    const v8i g1v = tdm_g1(KB, KB, Dh, Lh);        // V^T: 64x64 tile, row stride 2048

    if (issuer) {                                  // stage block 0 into buffer 0
        tdm_issue(tdm_g0((unsigned)(uintptr_t)&Kt[0][0][0], gaK), g1k);
        tdm_issue(tdm_g0((unsigned)(uintptr_t)&Vt[0][0][0], gaV), g1v);
        __builtin_amdgcn_s_wait_tensorcnt(0);
    }
    __syncthreads();

    for (int kb = 0; kb < nkb; ++kb) {
        const int cur = kb & 1;
        if (issuer && kb + 1 < nkb) {              // prefetch next block
            gaK += (unsigned long long)KB * Hh * Dh * 2;   // +128 KB
            gaV += (unsigned long long)KB * 2;             // +128 B
            tdm_issue(tdm_g0((unsigned)(uintptr_t)&Kt[cur ^ 1][0][0], gaK), g1k);
            tdm_issue(tdm_g0((unsigned)(uintptr_t)&Vt[cur ^ 1][0][0], gaV), g1v);
        }
        if (kb + 1 < nkb)                          // fully below diagonal
            process_block<false>(Kt[cur], Vt[cur], Pb[wave], aq, acc, m, l,
                                 kb * KB, qrow0, qrow0_s, ln16, hi);
        else                                       // diagonal block: masked
            process_block<true>(Kt[cur], Vt[cur], Pb[wave], aq, acc, m, l,
                                kb * KB, qrow0, qrow0_s, ln16, hi);
        if (issuer) __builtin_amdgcn_s_wait_tensorcnt(0);
        __syncthreads();                           // publish prefetched buffer
    }

    // epilogue: normalize, store [B,H,L,D]
    #pragma unroll
    for (int r = 0; r < 8; ++r) l[r] = 1.0f / l[r];
    float* ob = Out + (((size_t)b * Hh + h) * Lh) * Dh;
    #pragma unroll
    for (int j = 0; j < 4; ++j)
        #pragma unroll
        for (int r = 0; r < 8; ++r) {
            const int rowg = qrow0 + r + 8 * hi;
            ob[(size_t)rowg * Dh + 16 * j + ln16] = acc[j][r] * l[r];
        }
}

extern "C" void kernel_launch(void* const* d_in, const int* in_sizes, int n_in,
                              void* d_out, int out_size, void* d_ws, size_t ws_size,
                              hipStream_t stream) {
    const float* Q = (const float*)d_in[0];   // queries [B,L,H,D]
    const float* K = (const float*)d_in[1];   // keys    [B,L,H,D]
    const float* V = (const float*)d_in[2];   // values  [B,L,H,D]
    float* Out = (float*)d_out;               // [B,H,L,D]
    (void)in_sizes; (void)n_in; (void)out_size; (void)ws_size;

    const size_t N = (size_t)Bb * Lh * Hh * Dh;          // 8.4M elements
    unsigned* Kbf = (unsigned*)d_ws;                      // bf16 K, 16.8 MB
    unsigned* Vtb = (unsigned*)((char*)d_ws + N * 2);     // bf16 V^T, 16.8 MB

    cvt_k_bf16<<<dim3(N / 512), dim3(256), 0, stream>>>(K, Kbf);
    cvt_v_bf16_t<<<dim3(Bb * Hh * (Lh / 64)), dim3(256), 0, stream>>>(V, Vtb);

    dim3 grid(Lh / BQ, Bb * Hh);   // (32, 64)
    logsparse_flash_causal<<<grid, dim3(128), 0, stream>>>(
        Q, (const __bf16*)Kbf, (const __bf16*)Vtb, Out);
}
